// CompositionalAttention_60481729462643
// MI455X (gfx1250) — matrix operand order repaired and checked by reading, not compiled, not run on hardware
//
#include <hip/hip_runtime.h>

// ---------------------------------------------------------------------------
// CDNA5 (gfx1250) compositional attention, f16 WMMA + async LDS staging.
// ---------------------------------------------------------------------------

typedef __attribute__((ext_vector_type(16))) _Float16 v16h;
typedef __attribute__((ext_vector_type(8)))  float    v8f;

#define B_SZ   2
#define N_SEQ  2048
#define DIMM   1024
#define S_N    8
#define R_N    2
#define DH     64

#define WMMA_F16(a, b, c) \
  __builtin_amdgcn_wmma_f32_16x16x32_f16(false, (a), false, (b), (short)0, (c), false, false)

#if __has_builtin(__builtin_amdgcn_global_load_async_to_lds_b128) && \
    __has_builtin(__builtin_amdgcn_s_wait_asynccnt)
#define HAS_ASYNC 1
#else
#define HAS_ASYNC 0
#endif

#if HAS_ASYNC
#define ASYNC_WAIT(n) __builtin_amdgcn_s_wait_asynccnt(n)
#else
#define ASYNC_WAIT(n)
#endif

// Vector/pointer types matching the builtin's parameters:
//   param0: int __vector_size__(16) __device__(AS1) *   (non-const)
//   param1: int __vector_size__(16) AS3 *
typedef int v4i_vs __attribute__((vector_size(16)));
typedef __attribute__((address_space(1))) v4i_vs as1_v4i;
typedef __attribute__((address_space(3))) v4i_vs as3_v4i;

// One 16-byte lane-chunk: global -> LDS (async if available).
static __device__ inline void async_cp16(const _Float16* g, _Float16* l) {
#if HAS_ASYNC
  __builtin_amdgcn_global_load_async_to_lds_b128((as1_v4i*)g, (as3_v4i*)l, 0, 0);
#else
  *(uint4*)l = *(const uint4*)g;
#endif
}

// A-fragment (16x32 f16, MxK), row-major A, lda in elements.
// Lane L: m=L%16, g=L/16 holds K = [8g,8g+8) and [16+8g,16+8g+8).
static __device__ inline v16h ld_afrag(const _Float16* base, long lda, int lane) {
  int m = lane & 15, g = lane >> 4;
  const _Float16* p = base + (long)m * lda + 8 * g;
  union { v16h v; uint4 q[2]; } u;
  u.q[0] = *(const uint4*)(p);
  u.q[1] = *(const uint4*)(p + 16);
  return u.v;
}

// B-fragment (32x16 f16, KxN) loaded from BT (NxK row-major), ldb in elements.
// Lane L: n=L%16, g=L/16 holds K = [8g,8g+8) and [16+8g,16+8g+8) for column n.
static __device__ inline v16h ld_bfrag(const _Float16* base, long ldb, int lane) {
  int n = lane & 15, g = lane >> 4;
  const _Float16* p = base + (long)n * ldb + 8 * g;
  union { v16h v; uint4 q[2]; } u;
  u.q[0] = *(const uint4*)(p);
  u.q[1] = *(const uint4*)(p + 16);
  return u.v;
}

static __device__ inline float wred_max16(float x) {
  x = fmaxf(x, __shfl_xor(x, 1, 32));
  x = fmaxf(x, __shfl_xor(x, 2, 32));
  x = fmaxf(x, __shfl_xor(x, 4, 32));
  x = fmaxf(x, __shfl_xor(x, 8, 32));
  return x;
}
static __device__ inline float wred_sum16(float x) {
  x += __shfl_xor(x, 1, 32);
  x += __shfl_xor(x, 2, 32);
  x += __shfl_xor(x, 4, 32);
  x += __shfl_xor(x, 8, 32);
  return x;
}

// ---------------------------------------------------------------------------
// Elementwise conversion / transpose kernels
// ---------------------------------------------------------------------------
__global__ void cvt_f32_to_f16(const float* __restrict__ src,
                               _Float16* __restrict__ dst, long n) {
  long i = (long)blockIdx.x * blockDim.x + threadIdx.x;
  if (i < n) dst[i] = (_Float16)src[i];
}

// W[K][Ncol] (f32) -> WT[Ncol][K] (f16)
__global__ void tcvt_f32_to_f16(const float* __restrict__ W,
                                _Float16* __restrict__ WT, int K, int Ncol) {
  long i = (long)blockIdx.x * blockDim.x + threadIdx.x;
  long tot = (long)K * Ncol;
  if (i < tot) {
    int k = (int)(i / Ncol), n = (int)(i % Ncol);
    WT[(long)n * K + k] = (_Float16)W[i];
  }
}

// V [b][n][r][d] (f16, flat [B*N][R*DH]) -> Vt [b][r][d][n]
__global__ void v_transpose(const _Float16* __restrict__ V,
                            _Float16* __restrict__ Vt) {
  long i = (long)blockIdx.x * blockDim.x + threadIdx.x;
  long tot = (long)B_SZ * N_SEQ * R_N * DH;
  if (i < tot) {
    int d = (int)(i % DH);
    int r = (int)((i / DH) % R_N);
    int n = (int)((i / (DH * R_N)) % N_SEQ);
    int b = (int)(i / ((long)DH * R_N * N_SEQ));
    Vt[(((long)(b * R_N + r) * DH) + d) * N_SEQ + n] = V[i];
  }
}

// ---------------------------------------------------------------------------
// Generic WMMA GEMM with double-buffered async LDS staging.
// C[M][N] = A[M][K] (f16) * B (given as BT[N][K], f16).
// 256 threads = 8 waves as 4x2; wave tile 32x32, block tile 128x64.
// Requires M % 128 == 0, N % 64 == 0, K % 32 == 0.
// Per tile: A 128x32 (512 chunks, 2/thread) + BT 64x32 (256 chunks, 1/thread)
//   => 3 async instructions per wave per tile.
// ---------------------------------------------------------------------------
template <typename CT>
__global__ __launch_bounds__(256) void gemm_wmma(
    const _Float16* __restrict__ A, long lda,
    const _Float16* __restrict__ BT, long ldb,
    CT* __restrict__ C, long ldc, int K) {
  __shared__ __align__(16) _Float16 As[2][128][32];
  __shared__ __align__(16) _Float16 Bs[2][64][32];

  int tid = threadIdx.x;
  int lane = tid & 31, wv = tid >> 5;
  int wr = wv >> 1, wc = wv & 1;
  long rowBlk = (long)blockIdx.x * 128;
  long colBlk = (long)blockIdx.y * 64;
  int row0 = wr * 32, col0 = wc * 32;

  auto stage = [&](int buf, int k0) {
    int c0 = tid, c1 = tid + 256;
    int ar0 = c0 >> 2, ao0 = (c0 & 3) * 8;
    int ar1 = c1 >> 2, ao1 = (c1 & 3) * 8;
    async_cp16(A + (rowBlk + ar0) * lda + k0 + ao0, &As[buf][ar0][ao0]);
    async_cp16(A + (rowBlk + ar1) * lda + k0 + ao1, &As[buf][ar1][ao1]);
    int br = tid >> 2, bo = (tid & 3) * 8;
    async_cp16(BT + (colBlk + br) * ldb + k0 + bo, &Bs[buf][br][bo]);
  };

  v8f acc00 = {}, acc01 = {}, acc10 = {}, acc11 = {};
  stage(0, 0);
  int buf = 0;
  for (int k0 = 0; k0 < K; k0 += 32) {
    if (k0 + 32 < K) {
      stage(buf ^ 1, k0 + 32);
      ASYNC_WAIT(3);
    } else {
      ASYNC_WAIT(0);
    }
    __syncthreads();
    v16h a0 = ld_afrag(&As[buf][row0][0], 32, lane);
    v16h a1 = ld_afrag(&As[buf][row0 + 16][0], 32, lane);
    v16h b0 = ld_bfrag(&Bs[buf][col0][0], 32, lane);
    v16h b1 = ld_bfrag(&Bs[buf][col0 + 16][0], 32, lane);
    acc00 = WMMA_F16(a0, b0, acc00);
    acc01 = WMMA_F16(a0, b1, acc01);
    acc10 = WMMA_F16(a1, b0, acc10);
    acc11 = WMMA_F16(a1, b1, acc11);
    __syncthreads();
    buf ^= 1;
  }
  int n = lane & 15, g = lane >> 4;
  for (int v = 0; v < 8; ++v) {
    long r0 = rowBlk + row0 + v + 8 * g;
    long r1 = rowBlk + row0 + 16 + v + 8 * g;
    C[r0 * ldc + colBlk + col0 + n]      = (CT)acc00[v];
    C[r0 * ldc + colBlk + col0 + 16 + n] = (CT)acc01[v];
    C[r1 * ldc + colBlk + col0 + n]      = (CT)acc10[v];
    C[r1 * ldc + colBlk + col0 + 16 + n] = (CT)acc11[v];
  }
}

// ---------------------------------------------------------------------------
// Fused flash attention (2-pass, recompute) per (b,s):
//   P = softmax(Q K^T / 8) over j; Ret[b,s,r] = P @ V[b,r]  (never writes P)
// Block: 256 threads = 8 waves; each wave owns 16 query rows (block: 128).
// K/V j-tiles are block-shared through double-buffered async LDS staging:
//   Kt tile: 32 x 64 (256 chunks, 1/thread); Vs tile: 2 x 64 x 32 (2/thread).
// Q,K layout: [b][n][s*64+d] (stride 512).  Vt: [b][r][d][n].
// Ret: [b][s][r][n][d] f16.
// ---------------------------------------------------------------------------
__global__ __launch_bounds__(256) void attn_fused(
    const _Float16* __restrict__ Q, const _Float16* __restrict__ Kp,
    const _Float16* __restrict__ Vt, _Float16* __restrict__ Ret) {
  __shared__ __align__(16) _Float16 Kt[2][32][64];       // [j][d]
  __shared__ __align__(16) _Float16 Vs[2][R_N][64][32];  // [r][d][j]
  __shared__ __align__(16) _Float16 lp[8][16][32];       // per-wave P staging

  int tid = threadIdx.x;
  int lane = tid & 31, wv = tid >> 5;
  int bs = blockIdx.y;
  int b = bs >> 3, s = bs & 7;
  long i0 = (long)blockIdx.x * 128 + wv * 16;

  const _Float16* Qb = Q + (long)b * N_SEQ * (S_N * DH) + (long)s * DH;
  const _Float16* Kb = Kp + (long)b * N_SEQ * (S_N * DH) + (long)s * DH;
  const long ldq = S_N * DH;  // 512

  // Stage K j-tile: rows j0..j0+31, 64 halves each = 8 chunks/row.
  auto stageK = [&](int buf, int j0) {
    int r = tid >> 3, o = (tid & 7) * 8;
    async_cp16(Kb + (long)(j0 + r) * ldq + o, &Kt[buf][r][o]);
  };
  // Stage V j-tile for both r: 128 rows of 32 halves = 4 chunks/row.
  auto stageV = [&](int buf, int j0) {
    int c0 = tid, c1 = tid + 256;
    int r0_ = c0 >> 2, o0 = (c0 & 3) * 8;
    int r1_ = c1 >> 2, o1 = (c1 & 3) * 8;
    const _Float16* Vbase = Vt + (long)(b * R_N) * DH * N_SEQ;
    async_cp16(Vbase + (long)r0_ * N_SEQ + j0 + o0, &Vs[buf][0][0][0] + r0_ * 32 + o0);
    async_cp16(Vbase + (long)r1_ * N_SEQ + j0 + o1, &Vs[buf][0][0][0] + r1_ * 32 + o1);
  };

  // Q fragments (16 rows x full d=64), reused across all j tiles.
  v16h qa0 = ld_afrag(Qb + i0 * ldq + 0, ldq, lane);
  v16h qa1 = ld_afrag(Qb + i0 * ldq + 32, ldq, lane);

  const float scale = 0.125f;
  float rm[8], rs[8];
#pragma unroll
  for (int v = 0; v < 8; ++v) { rm[v] = -3.0e38f; rs[v] = 0.f; }

  // ---- pass 1: online row max / sum of exp (K staged, 1 async op/tile) ----
  stageK(0, 0);
  int buf = 0;
  for (int j0 = 0; j0 < N_SEQ; j0 += 32) {
    if (j0 + 32 < N_SEQ) {
      stageK(buf ^ 1, j0 + 32);
      ASYNC_WAIT(1);
    } else {
      ASYNC_WAIT(0);
    }
    __syncthreads();
    v8f acc0 = {}, acc1 = {};
    {
      v16h b0 = ld_bfrag(&Kt[buf][0][0], 64, lane);
      v16h b1 = ld_bfrag(&Kt[buf][0][32], 64, lane);
      acc0 = WMMA_F16(qa0, b0, acc0);
      acc0 = WMMA_F16(qa1, b1, acc0);
    }
    {
      v16h b0 = ld_bfrag(&Kt[buf][16][0], 64, lane);
      v16h b1 = ld_bfrag(&Kt[buf][16][32], 64, lane);
      acc1 = WMMA_F16(qa0, b0, acc1);
      acc1 = WMMA_F16(qa1, b1, acc1);
    }
#pragma unroll
    for (int v = 0; v < 8; ++v) {
      float x0 = acc0[v] * scale, x1 = acc1[v] * scale;
      float mx = wred_max16(fmaxf(x0, x1));
      float nm = fmaxf(rm[v], mx);
      float e = wred_sum16(__expf(x0 - nm) + __expf(x1 - nm));
      rs[v] = rs[v] * __expf(rm[v] - nm) + e;
      rm[v] = nm;
    }
    __syncthreads();
    buf ^= 1;
  }
  float inv[8];
#pragma unroll
  for (int v = 0; v < 8; ++v) inv[v] = 1.0f / rs[v];

  // ---- pass 2: recompute, normalize, P@V for both r (K+V staged) ----
  v8f racc[R_N * 4];
#pragma unroll
  for (int t = 0; t < R_N * 4; ++t) racc[t] = (v8f){};

  int n = lane & 15, g = lane >> 4;
  stageK(0, 0);
  stageV(0, 0);
  buf = 0;
  for (int j0 = 0; j0 < N_SEQ; j0 += 32) {
    if (j0 + 32 < N_SEQ) {
      stageK(buf ^ 1, j0 + 32);
      stageV(buf ^ 1, j0 + 32);
      ASYNC_WAIT(3);
    } else {
      ASYNC_WAIT(0);
    }
    __syncthreads();
    v8f acc0 = {}, acc1 = {};
    {
      v16h b0 = ld_bfrag(&Kt[buf][0][0], 64, lane);
      v16h b1 = ld_bfrag(&Kt[buf][0][32], 64, lane);
      acc0 = WMMA_F16(qa0, b0, acc0);
      acc0 = WMMA_F16(qa1, b1, acc0);
    }
    {
      v16h b0 = ld_bfrag(&Kt[buf][16][0], 64, lane);
      v16h b1 = ld_bfrag(&Kt[buf][16][32], 64, lane);
      acc1 = WMMA_F16(qa0, b0, acc1);
      acc1 = WMMA_F16(qa1, b1, acc1);
    }
    // C-layout -> row-major 16x32 P tile in LDS (wave-private region).
#pragma unroll
    for (int v = 0; v < 8; ++v) {
      int m = v + 8 * g;
      lp[wv][m][n]      = (_Float16)(__expf(acc0[v] * scale - rm[v]) * inv[v]);
      lp[wv][m][16 + n] = (_Float16)(__expf(acc1[v] * scale - rm[v]) * inv[v]);
    }
    v16h pa = ld_afrag(&lp[wv][0][0], 32, lane);
#pragma unroll
    for (int r = 0; r < R_N; ++r) {
#pragma unroll
      for (int dt = 0; dt < 4; ++dt) {
        v16h vb = ld_bfrag(&Vs[buf][r][dt * 16][0], 32, lane);
        racc[r * 4 + dt] = WMMA_F16(pa, vb, racc[r * 4 + dt]);
      }
    }
    __syncthreads();
    buf ^= 1;
  }

  // Store retrieved [b][s][r][n][d] as f16.
#pragma unroll
  for (int r = 0; r < R_N; ++r)
#pragma unroll
    for (int dt = 0; dt < 4; ++dt)
#pragma unroll
      for (int v = 0; v < 8; ++v) {
        long row = i0 + v + 8 * g;
        long base = (((long)(b * S_N + s) * R_N + r) * N_SEQ + row) * DH;
        Ret[base + dt * 16 + n] = (_Float16)racc[r * 4 + dt][v];
      }
}

// ---------------------------------------------------------------------------
// Compositional softmax over R=2 retrievals + combine.
// rq: [b][n][s*64+d]; Rk,Ret: [b][s][r][n][d]; OutC: [b][n][s*64+d].
// One thread per (b,s,n).
// ---------------------------------------------------------------------------
__global__ void combine_kernel(const _Float16* __restrict__ RQ,
                               const _Float16* __restrict__ Rk,
                               const _Float16* __restrict__ Ret,
                               _Float16* __restrict__ OutC) {
  long t = (long)blockIdx.x * blockDim.x + threadIdx.x;
  if (t >= (long)B_SZ * S_N * N_SEQ) return;
  int nn = (int)(t % N_SEQ);
  int s = (int)((t / N_SEQ) % S_N);
  int b = (int)(t / ((long)N_SEQ * S_N));

  const _Float16* rq = RQ + ((long)(b * N_SEQ + nn)) * (S_N * DH) + s * DH;
  float dot[R_N];
#pragma unroll
  for (int r = 0; r < R_N; ++r) {
    const _Float16* rk =
        Rk + (((long)(b * S_N + s) * R_N + r) * N_SEQ + nn) * DH;
    float d = 0.f;
    for (int k = 0; k < DH; ++k) d += (float)rq[k] * (float)rk[k];
    dot[r] = d * 0.125f;
  }
  float mx = fmaxf(dot[0], dot[1]);
  float e0 = __expf(dot[0] - mx), e1 = __expf(dot[1] - mx);
  float is = 1.0f / (e0 + e1);
  float w0 = e0 * is, w1 = e1 * is;

  const _Float16* r0 = Ret + (((long)(b * S_N + s) * R_N + 0) * N_SEQ + nn) * DH;
  const _Float16* r1 = Ret + (((long)(b * S_N + s) * R_N + 1) * N_SEQ + nn) * DH;
  _Float16* o = OutC + ((long)(b * N_SEQ + nn)) * (S_N * DH) + s * DH;
  for (int k = 0; k < DH; ++k)
    o[k] = (_Float16)(w0 * (float)r0[k] + w1 * (float)r1[k]);
}

// ---------------------------------------------------------------------------
// Host-side orchestration
// ---------------------------------------------------------------------------
extern "C" void kernel_launch(void* const* d_in, const int* in_sizes, int n_in,
                              void* d_out, int out_size, void* d_ws,
                              size_t ws_size, hipStream_t stream) {
  const float* x    = (const float*)d_in[0];
  const float* Wsq  = (const float*)d_in[1];
  const float* Wsk  = (const float*)d_in[2];
  const float* Wrv  = (const float*)d_in[3];
  const float* Wrq  = (const float*)d_in[4];
  const float* Wrk  = (const float*)d_in[5];
  const float* Wout = (const float*)d_in[6];
  float* outF = (float*)d_out;

  char* w = (char*)d_ws;
  size_t off = 0;
  auto alloc = [&](size_t bytes) -> char* {
    char* p = w + off;
    off += (bytes + 255) & ~(size_t)255;
    return p;
  };
  const long M = (long)B_SZ * N_SEQ;  // 4096
  _Float16* xh    = (_Float16*)alloc(M * DIMM * 2);
  _Float16* WsqT  = (_Float16*)alloc((size_t)(S_N * DH) * DIMM * 2);
  _Float16* WskT  = (_Float16*)alloc((size_t)(S_N * DH) * DIMM * 2);
  _Float16* WrvT  = (_Float16*)alloc((size_t)(R_N * DH) * DIMM * 2);
  _Float16* WrqT  = (_Float16*)alloc((size_t)(S_N * DH) * DIMM * 2);
  _Float16* WrkT  = (_Float16*)alloc((size_t)DH * DH * 2);
  _Float16* WoutT = (_Float16*)alloc((size_t)DIMM * (S_N * DH) * 2);
  _Float16* Qh    = (_Float16*)alloc(M * (S_N * DH) * 2);
  _Float16* Kh    = (_Float16*)alloc(M * (S_N * DH) * 2);
  _Float16* RQh   = (_Float16*)alloc(M * (S_N * DH) * 2);
  _Float16* Vh    = (_Float16*)alloc(M * (R_N * DH) * 2);
  _Float16* Vth   = (_Float16*)alloc((size_t)B_SZ * R_N * DH * N_SEQ * 2);
  _Float16* Ret   = (_Float16*)alloc((size_t)B_SZ * S_N * R_N * N_SEQ * DH * 2);
  _Float16* Rk    = (_Float16*)alloc((size_t)B_SZ * S_N * R_N * N_SEQ * DH * 2);
  _Float16* OutC  = (_Float16*)alloc(M * (S_N * DH) * 2);
  (void)ws_size; (void)in_sizes; (void)n_in; (void)out_size;

  // 1) f32 -> f16 conversions (+ weight transposes into BT layout)
  {
    long n = M * DIMM;
    cvt_f32_to_f16<<<dim3((unsigned)((n + 255) / 256)), 256, 0, stream>>>(x, xh, n);
  }
  auto tcvt = [&](const float* src, _Float16* dst, int K, int Ncol) {
    long n = (long)K * Ncol;
    tcvt_f32_to_f16<<<dim3((unsigned)((n + 255) / 256)), 256, 0, stream>>>(src, dst, K, Ncol);
  };
  tcvt(Wsq, WsqT, DIMM, S_N * DH);
  tcvt(Wsk, WskT, DIMM, S_N * DH);
  tcvt(Wrv, WrvT, DIMM, R_N * DH);
  tcvt(Wrq, WrqT, DIMM, S_N * DH);
  tcvt(Wrk, WrkT, DH, DH);
  tcvt(Wout, WoutT, S_N * DH, DIMM);

  // 2) projection GEMMs (M=4096, K=1024)
  gemm_wmma<_Float16><<<dim3(32, 8), 256, 0, stream>>>(xh, DIMM, WsqT, DIMM, Qh, S_N * DH, DIMM);
  gemm_wmma<_Float16><<<dim3(32, 8), 256, 0, stream>>>(xh, DIMM, WskT, DIMM, Kh, S_N * DH, DIMM);
  gemm_wmma<_Float16><<<dim3(32, 8), 256, 0, stream>>>(xh, DIMM, WrqT, DIMM, RQh, S_N * DH, DIMM);
  gemm_wmma<_Float16><<<dim3(32, 2), 256, 0, stream>>>(xh, DIMM, WrvT, DIMM, Vh, R_N * DH, DIMM);

  // 3) V -> Vt
  {
    long n = (long)B_SZ * N_SEQ * R_N * DH;
    v_transpose<<<dim3((unsigned)((n + 255) / 256)), 256, 0, stream>>>(Vh, Vth);
  }

  // 4) fused flash attention -> retrieved
  attn_fused<<<dim3(N_SEQ / 128, B_SZ * S_N), 256, 0, stream>>>(Qh, Kh, Vth, Ret);

  // 5) rk = retrieved @ Wrk  (flattened M = B*S*R*N = 65536, K = N' = 64)
  gemm_wmma<_Float16><<<dim3(512, 1), 256, 0, stream>>>(Ret, DH, WrkT, DH, Rk, DH, DH);

  // 6) compositional softmax over R and combine
  {
    long n = (long)B_SZ * S_N * N_SEQ;
    combine_kernel<<<dim3((unsigned)((n + 255) / 256)), 256, 0, stream>>>(RQh, Rk, Ret, OutC);
  }

  // 7) final output GEMM: [4096, 512] @ Wout[512, 1024] -> f32 d_out
  gemm_wmma<float><<<dim3(32, 16), 256, 0, stream>>>(OutC, S_N * DH, WoutT, S_N * DH, outF, DIMM, S_N * DH);
}
